// MultiHeadPiiModel_11982958756384
// MI455X (gfx1250) — matrix operand order repaired and checked by reading, not compiled, not run on hardware
//
#include <hip/hip_runtime.h>
#include <hip/hip_bf16.h>
#include <math.h>
#include <stdint.h>

// ---------------------------------------------------------------------------
// MultiHeadPiiModel on MI455X (gfx1250, wave32)
//   - big MLP GEMMs: v_wmma_f32_16x16x32_f16, 128x128 block tile, double-
//     buffered LDS fed by global_load_async_to_lds_b128 (inline asm; ASYNCcnt)
//   - W pre-transposed to [N][K] f16 so all tile feeds are b128 copies
//   - skinny heads (3/18/4 outputs) as wave-per-row VALU dots (mem-bound)
// ---------------------------------------------------------------------------

typedef __attribute__((ext_vector_type(16))) _Float16 v16h;
typedef __attribute__((ext_vector_type(8)))  _Float16 v8h;
typedef __attribute__((ext_vector_type(8)))  float    v8f;

#define BQ 8
#define SQ 2048
#define HQ 768
#define NSPAN 512
#define NROW (BQ * NSPAN)      // 4096 span rows
#define SPAN_DIM 2688          // 3*H + H/2
#define KPAD 2720              // 2688 + 18 (type probs) + 14 zero pad; 170*16
#define NTYPE 18
#define NSENS 4

// ---- CDNA5 async global->LDS path (asm load + s_wait_asynccnt builtin) ----
#if defined(__has_builtin)
#if __has_builtin(__builtin_amdgcn_s_wait_asynccnt)
#define USE_ASYNC_COPY 1
#endif
#endif
#ifndef USE_ASYNC_COPY
#define USE_ASYNC_COPY 0
#endif

__device__ __forceinline__ void cp_b128(const _Float16* gsrc, _Float16* ldst) {
#if USE_ASYNC_COPY
    // Generic LDS pointers are {aperture | offset}; low 32 bits are the LDS
    // byte address (ISA 10.2). Each lane copies its own 16B chunk.
    unsigned loff = (unsigned)(size_t)ldst;
    asm volatile("global_load_async_to_lds_b128 %0, %1, off"
                 :: "v"(loff), "v"(gsrc)
                 : "memory");
#else
    *(v8h*)ldst = *(const v8h*)gsrc;
#endif
}

template <int N>
__device__ __forceinline__ void wait_async_le() {
#if USE_ASYNC_COPY
    __builtin_amdgcn_s_wait_asynccnt(N);
#endif
}

__device__ __forceinline__ float gelu_exact(float x) {
    return 0.5f * x * (1.0f + erff(x * 0.70710678118654752f));
}

// ---------------- W [K][N] f32 -> WT [N][KP] f16 (tiled transpose) ---------
__global__ void transpose_cvt(const float* __restrict__ src,
                              _Float16* __restrict__ dst,
                              int K, int KP) {
    __shared__ float tile[32][33];
    const int kt = blockIdx.x * 32;          // k tile base (covers KP exactly)
    const int nt = blockIdx.y * 32;          // n tile base (N = 768)
    const int tx = threadIdx.x & 31;
    const int ty = threadIdx.x >> 5;         // 0..7
    #pragma unroll
    for (int r = ty; r < 32; r += 8) {
        int k = kt + r;
        tile[r][tx] = (k < K) ? src[(size_t)k * HQ + nt + tx] : 0.0f;
    }
    __syncthreads();
    #pragma unroll
    for (int r = ty; r < 32; r += 8) {
        dst[(size_t)(nt + r) * KP + kt + tx] = (_Float16)tile[tx][r];
    }
}

// ---------------- proposal head: hidden[16384,768] @ W_prop[768,3] ---------
__global__ void prop_head(const float* __restrict__ hidden,
                          const float* __restrict__ Wp,
                          const float* __restrict__ bp,
                          float* __restrict__ out) {
    int gw   = (blockIdx.x * blockDim.x + threadIdx.x) >> 5;  // wave per row
    int lane = threadIdx.x & 31;
    if (gw >= BQ * SQ) return;
    const float* h = hidden + (size_t)gw * HQ;
    float a0 = 0.f, a1 = 0.f, a2 = 0.f;
    for (int k = lane; k < HQ; k += 32) {
        float v = h[k];
        a0 += v * Wp[k * 3 + 0];
        a1 += v * Wp[k * 3 + 1];
        a2 += v * Wp[k * 3 + 2];
    }
    for (int off = 16; off; off >>= 1) {
        a0 += __shfl_xor(a0, off, 32);
        a1 += __shfl_xor(a1, off, 32);
        a2 += __shfl_xor(a2, off, 32);
    }
    if (lane == 0) {
        out[gw * 3 + 0] = a0 + bp[0];
        out[gw * 3 + 1] = a1 + bp[1];
        out[gw * 3 + 2] = a2 + bp[2];
    }
}

// ---------------- span representations, written as f16 GEMM-A --------------
__global__ void span_reprs(const float* __restrict__ hidden,
                           const int* __restrict__ spans,
                           const float* __restrict__ wemb,
                           _Float16* __restrict__ rep) {
    int row = blockIdx.x;           // 0..4095
    int b   = row >> 9;             // row / 512
    int t   = threadIdx.x;          // 256 threads
    int s = spans[row * 2 + 0];
    int e = spans[row * 2 + 1];
    int w = e - s + 1;
    const float* hb = hidden + (size_t)b * SQ * HQ;
    _Float16* r = rep + (size_t)row * KPAD;
    float inv_w = 1.0f / (float)w;
    for (int c = t; c < HQ; c += 256) {
        float sh = hb[(size_t)s * HQ + c];
        float eh = hb[(size_t)e * HQ + c];
        float acc = 0.f;
        for (int q = s; q <= e; ++q) acc += hb[(size_t)q * HQ + c];
        r[c]          = (_Float16)sh;
        r[HQ + c]     = (_Float16)eh;
        r[2 * HQ + c] = (_Float16)(acc * inv_w);
    }
    int wc = (w < 63) ? w : 63;
    for (int c = t; c < HQ / 2; c += 256)
        r[3 * HQ + c] = (_Float16)wemb[wc * (HQ / 2) + c];
    if (t < KPAD - SPAN_DIM)                       // zero padding tail (32 cols)
        r[SPAN_DIM + t] = (_Float16)0.0f;
}

// ---------------- tiled WMMA GEMM + bias + exact GELU ----------------------
// C[4096,768] = gelu(A[4096,K](f16, lda=KPAD) @ WT[768,K]^T (f16) + bias)
// block tile 128x128; 8 waves as 2(M) x 4(N); wave tile 64x32 = 4x2 WMMA
// double-buffered LDS, async global->LDS feed
__global__ void __launch_bounds__(256)
gemm_gelu_wmma(const _Float16* __restrict__ A, int K,
               const _Float16* __restrict__ WT,   // [768][K], stride K
               const float* __restrict__ bias,
               float* __restrict__ out) {
    __shared__ __align__(32) _Float16 sA[2][128 * 32];   // [m][k], 8 KB each
    __shared__ __align__(32) _Float16 sB[2][128 * 32];   // [n][k], 8 KB each

    const int tid   = threadIdx.x;
    const int lane  = tid & 31;
    const int wave  = tid >> 5;
    const int waveM = wave >> 2;        // 0..1
    const int waveN = wave & 3;         // 0..3
    const int m0 = blockIdx.x * 128;
    const int n0 = blockIdx.y * 128;
    const int wm = waveM * 64, wn = waveN * 32;
    const int lm  = lane & 15;
    const int kbA = (lane < 16) ? 0 : 16;   // A frag: lane half h takes k 16h..16h+15
    const int kbB = (lane < 16) ? 0 : 16;   // B frag: lane half h takes k 16h..16h+15

    // per-thread tile feed: 2 b128 chunks for A + 2 for B
    const int c0r = tid >> 2, c0q = tid & 3;            // chunk tid
    const int c1r = (tid + 256) >> 2, c1q = tid & 3;    // chunk tid+256

    auto issue_tile = [&](int kc, int buf) {
        cp_b128(A + (size_t)(m0 + c0r) * KPAD + kc + c0q * 8,
                &sA[buf][c0r * 32 + c0q * 8]);
        cp_b128(A + (size_t)(m0 + c1r) * KPAD + kc + c1q * 8,
                &sA[buf][c1r * 32 + c1q * 8]);
        cp_b128(WT + (size_t)(n0 + c0r) * K + kc + c0q * 8,
                &sB[buf][c0r * 32 + c0q * 8]);
        cp_b128(WT + (size_t)(n0 + c1r) * K + kc + c1q * 8,
                &sB[buf][c1r * 32 + c1q * 8]);
    };

    v8f acc[4][2] = {};
    const int iters = K / 32;
    issue_tile(0, 0);

    for (int i = 0; i < iters; ++i) {
        const int buf = i & 1;
        const bool more = (i + 1 < iters);
        if (more) {
            issue_tile((i + 1) * 32, buf ^ 1);   // prefetch next tile
            wait_async_le<4>();                  // current tile's 4 chunks done
        } else {
            wait_async_le<0>();                  // drain
        }
        __syncthreads();

        v16h af[4], bf[2];
        #pragma unroll
        for (int mt = 0; mt < 4; ++mt) {
            int r = wm + mt * 16 + lm;
            af[mt] = *(const v16h*)(&sA[buf][r * 32 + kbA]);       // 16 contig K
        }
        #pragma unroll
        for (int nt = 0; nt < 2; ++nt) {
            int n = wn + nt * 16 + lm;
            bf[nt] = *(const v16h*)(&sB[buf][n * 32 + kbB]);       // 16 contig K
        }
        #pragma unroll
        for (int mt = 0; mt < 4; ++mt)
            #pragma unroll
            for (int nt = 0; nt < 2; ++nt)
                acc[mt][nt] = __builtin_amdgcn_wmma_f32_16x16x32_f16(
                    false, af[mt], false, bf[nt],
                    (short)0, acc[mt][nt], false, false);

        __syncthreads();   // all reads done before buf is overwritten next iter
    }

    // epilogue: bias + exact GELU, f32 out
    const int mhi = (lane & 16) ? 8 : 0;   // C layout: lanes 16-31 hold M+8
    #pragma unroll
    for (int mt = 0; mt < 4; ++mt) {
        #pragma unroll
        for (int nt = 0; nt < 2; ++nt) {
            int gc = n0 + wn + nt * 16 + lm;
            float bv = bias[gc];
            #pragma unroll
            for (int r8 = 0; r8 < 8; ++r8) {
                int gr = m0 + wm + mt * 16 + r8 + mhi;
                out[(size_t)gr * HQ + gc] = gelu_exact(acc[mt][nt][r8] + bv);
            }
        }
    }
}

// ---------------- type head: t @ W_t2[768,18] + softmax -> rep cols --------
__global__ void type_head(const float* __restrict__ T,
                          const float* __restrict__ W2,
                          const float* __restrict__ b2,
                          float* __restrict__ logits_out,
                          _Float16* __restrict__ rep) {
    int gw   = (blockIdx.x * blockDim.x + threadIdx.x) >> 5;
    int lane = threadIdx.x & 31;
    if (gw >= NROW) return;
    const float* trow = T + (size_t)gw * HQ;
    float acc[NTYPE];
    #pragma unroll
    for (int j = 0; j < NTYPE; ++j) acc[j] = 0.f;
    for (int k = lane; k < HQ; k += 32) {
        float v = trow[k];
        #pragma unroll
        for (int j = 0; j < NTYPE; ++j) acc[j] += v * W2[k * NTYPE + j];
    }
    #pragma unroll
    for (int j = 0; j < NTYPE; ++j)
        for (int off = 16; off; off >>= 1) acc[j] += __shfl_xor(acc[j], off, 32);
    #pragma unroll
    for (int j = 0; j < NTYPE; ++j) acc[j] += b2[j];
    // softmax (redundant per lane; trivially cheap)
    float mx = acc[0];
    #pragma unroll
    for (int j = 1; j < NTYPE; ++j) mx = fmaxf(mx, acc[j]);
    float ex[NTYPE], sum = 0.f;
    #pragma unroll
    for (int j = 0; j < NTYPE; ++j) { ex[j] = __expf(acc[j] - mx); sum += ex[j]; }
    float inv = 1.0f / sum;
    if (lane == 0) {
        #pragma unroll
        for (int j = 0; j < NTYPE; ++j) {
            logits_out[(size_t)gw * NTYPE + j] = acc[j];
            rep[(size_t)gw * KPAD + SPAN_DIM + j] = (_Float16)(ex[j] * inv);
        }
    }
}

// ---------------- sensitivity head: s @ W_s2[768,4] ------------------------
__global__ void sens_head(const float* __restrict__ S,
                          const float* __restrict__ W2,
                          const float* __restrict__ b2,
                          float* __restrict__ out) {
    int gw   = (blockIdx.x * blockDim.x + threadIdx.x) >> 5;
    int lane = threadIdx.x & 31;
    if (gw >= NROW) return;
    const float* srow = S + (size_t)gw * HQ;
    float acc[NSENS];
    #pragma unroll
    for (int j = 0; j < NSENS; ++j) acc[j] = 0.f;
    for (int k = lane; k < HQ; k += 32) {
        float v = srow[k];
        #pragma unroll
        for (int j = 0; j < NSENS; ++j) acc[j] += v * W2[k * NSENS + j];
    }
    #pragma unroll
    for (int j = 0; j < NSENS; ++j)
        for (int off = 16; off; off >>= 1) acc[j] += __shfl_xor(acc[j], off, 32);
    if (lane == 0) {
        #pragma unroll
        for (int j = 0; j < NSENS; ++j)
            out[(size_t)gw * NSENS + j] = acc[j] + b2[j];
    }
}

// ---------------------------------------------------------------------------
extern "C" void kernel_launch(void* const* d_in, const int* in_sizes, int n_in,
                              void* d_out, int out_size, void* d_ws, size_t ws_size,
                              hipStream_t stream) {
    const float* hidden = (const float*)d_in[0];
    const int*   spans  = (const int*)d_in[1];
    const float* wemb   = (const float*)d_in[2];
    const float* Wp     = (const float*)d_in[3];
    const float* bp     = (const float*)d_in[4];
    const float* Wt1    = (const float*)d_in[5];
    const float* bt1    = (const float*)d_in[6];
    const float* Wt2    = (const float*)d_in[7];
    const float* bt2    = (const float*)d_in[8];
    const float* Ws1    = (const float*)d_in[9];
    const float* bs1    = (const float*)d_in[10];
    const float* Ws2    = (const float*)d_in[11];
    const float* bs2    = (const float*)d_in[12];

    // workspace carve-out (~56 MB)
    char*  ws  = (char*)d_ws;
    size_t off = 0;
    auto carve = [&](size_t bytes) -> void* {
        off = (off + 255) & ~(size_t)255;
        void* p = ws + off;
        off += bytes;
        return p;
    };
    _Float16* repH  = (_Float16*)carve((size_t)NROW * KPAD * 2);     // 22.3 MB
    _Float16* Wt1T  = (_Float16*)carve((size_t)HQ * SPAN_DIM * 2);   //  4.1 MB
    _Float16* Ws1T  = (_Float16*)carve((size_t)HQ * KPAD * 2);       //  4.2 MB
    float*    Tm    = (float*)carve((size_t)NROW * HQ * 4);          // 12.6 MB
    float*    Sm    = (float*)carve((size_t)NROW * HQ * 4);          // 12.6 MB

    float* out_prop = (float*)d_out;                  // [8,2048,3]
    float* out_type = out_prop + BQ * SQ * 3;         // [8,512,18]
    float* out_sens = out_type + NROW * NTYPE;        // [8,512,4]

    // 1) weight transpose + f16 conversion: W[K,768] -> WT[768,Kpad]
    transpose_cvt<<<dim3(SPAN_DIM / 32, HQ / 32), 256, 0, stream>>>(
        Wt1, Wt1T, SPAN_DIM, SPAN_DIM);
    transpose_cvt<<<dim3(KPAD / 32, HQ / 32), 256, 0, stream>>>(
        Ws1, Ws1T, SPAN_DIM + NTYPE, KPAD);
    // 2) proposal head (mem-bound over 48 MB hidden)
    prop_head<<<(BQ * SQ) / 8, 256, 0, stream>>>(hidden, Wp, bp, out_prop);
    // 3) span features -> f16 GEMM-A buffer
    span_reprs<<<NROW, 256, 0, stream>>>(hidden, spans, wemb, repH);
    // 4) t = gelu(rep @ W_t1 + b_t1)     (WMMA, K=2688)
    gemm_gelu_wmma<<<dim3(NROW / 128, HQ / 128), 256, 0, stream>>>(
        repH, SPAN_DIM, Wt1T, bt1, Tm);
    // 5) type logits + softmax; probs appended into rep cols 2688..2705
    type_head<<<NROW / 8, 256, 0, stream>>>(Tm, Wt2, bt2, out_type, repH);
    // 6) s = gelu(s_in @ W_s1 + b_s1)    (WMMA, K=2720 incl. zero pad)
    gemm_gelu_wmma<<<dim3(NROW / 128, HQ / 128), 256, 0, stream>>>(
        repH, KPAD, Ws1T, bs1, Sm);
    // 7) sensitivity head
    sens_head<<<NROW / 8, 256, 0, stream>>>(Sm, Ws2, bs2, out_sens);
}
